// UnifiedMoEAttention_62380105007281
// MI455X (gfx1250) — matrix operand order repaired and checked by reading, not compiled, not run on hardware
//
#include <hip/hip_runtime.h>
#include <hip/hip_bf16.h>

typedef __bf16 bf16;
typedef __attribute__((ext_vector_type(16))) __bf16 v16bf;
typedef __attribute__((ext_vector_type(8)))  __bf16 v8bf;
typedef __attribute__((ext_vector_type(8)))  float  v8f;
typedef __attribute__((ext_vector_type(4)))  int    v4i;
typedef __attribute__((ext_vector_type(8)))  int    v8i;
typedef __attribute__((ext_vector_type(4)))  unsigned int v4u;

static constexpr int B_  = 2;
static constexpr int S_  = 2048;
static constexpr int D_  = 1024;
static constexpr int H_  = 16;
static constexpr int HD_ = 64;
static constexpr int M_  = B_ * S_;    // 4096 rows of x
static constexpr int E_  = H_ * HD_;   // 1024

// ---------------------------------------------------------------------------
// CDNA5 async global->LDS copy (ASYNCcnt) + TDM (TENSORcnt), with fallbacks
// ---------------------------------------------------------------------------
#if defined(__has_builtin)
#if __has_builtin(__builtin_amdgcn_global_load_async_to_lds_b128)
#define HAVE_ASYNC_COPY 1
#endif
#if __has_builtin(__builtin_amdgcn_s_wait_asynccnt)
#define HAVE_WAIT_ASYNC 1
#endif
#if __has_builtin(__builtin_amdgcn_tensor_load_to_lds) && \
    __has_builtin(__builtin_amdgcn_s_wait_tensorcnt)
#define HAVE_TDM 1
#endif
#endif

typedef __attribute__((address_space(1))) v4i v4i_g;  // global int4
typedef __attribute__((address_space(3))) v4i v4i_l;  // LDS int4

__device__ inline void cp16(bf16* lds_dst, const bf16* gsrc) {
#ifdef HAVE_ASYNC_COPY
  __builtin_amdgcn_global_load_async_to_lds_b128(
      (v4i_g*)(__SIZE_TYPE__)(gsrc),   // AS1 bits == generic bits for global
      (v4i_l*)(lds_dst),               // addrspacecast generic -> AS3
      0, 0);
#else
  *reinterpret_cast<v8bf*>(lds_dst) = *reinterpret_cast<const v8bf*>(gsrc);
#endif
}

__device__ inline void wait_async0() {
#ifdef HAVE_ASYNC_COPY
#ifdef HAVE_WAIT_ASYNC
  __builtin_amdgcn_s_wait_asynccnt(0);
#else
  asm volatile("s_wait_asynccnt 0x0" ::: "memory");
#endif
#endif
}

#ifdef HAVE_TDM
__device__ inline unsigned lds_off_of(const void* p) {
  return (unsigned)(unsigned long long)
         (__attribute__((address_space(3))) const char*)p;
}

// TDM: DMA a 2-D tile (tile_dim0 x tile_dim1 elems of 2B, x contiguous with
// row stride `stride0` elems) starting at gsrc into LDS at lds byte offset.
// Rows land densely packed (tile_dim0*2 bytes apart) in LDS.
__device__ inline void tdm_load_2d(const bf16* gsrc, unsigned ldsb,
                                   unsigned dim0, unsigned stride0,
                                   unsigned dim1, unsigned t0, unsigned t1) {
  const unsigned long long ga = (unsigned long long)(__SIZE_TYPE__)gsrc;
  v4u g0 = {1u,                                   // count=1 (valid D#)
            ldsb,                                 // lds_addr (bytes)
            (unsigned)(ga & 0xffffffffu),         // global_addr[31:0]
            (unsigned)((ga >> 32) & 0x01ffffffu)  // global_addr[56:32]
              | 0x80000000u};                     // type=2 ("image")
  v8i g1 = {(int)0x00010000,                      // data_size=1 -> 2 bytes
            (int)((dim0 & 0xffffu) << 16),        // tensor_dim0[15:0]
            (int)((dim0 >> 16) | ((dim1 & 0xffffu) << 16)),
            (int)((dim1 >> 16) | (t0 << 16)),     // tile_dim0
            (int)(t1),                            // tile_dim1 (tile_dim2=0)
            (int)(stride0),                       // tensor_dim0_stride[31:0]
            0, 0};
  v4i gz4 = {0, 0, 0, 0};
  v8i gz8 = {0, 0, 0, 0, 0, 0, 0, 0};
  // 6-arg form (amdgpu-toolchain / clang-23 profile)
  __builtin_amdgcn_tensor_load_to_lds(g0, g1, gz4, gz4, gz8, 0);
}
#endif

// ---------------------------------------------------------------------------
// WMMA helpers (CDNA5 wave32, v_wmma_f32_16x16x32_bf16)
// ---------------------------------------------------------------------------
__device__ inline v8f wmma_bf16(v16bf a, v16bf b, v8f c) {
  return __builtin_amdgcn_wmma_f32_16x16x32_bf16(
      /*neg_a=*/false, a, /*neg_b=*/false, b,
      /*c_mod=*/(short)0, c, /*reuse_a=*/false, /*reuse_b=*/false);
}

// A fragment, 16x32 bf16, memory row-major (K contiguous, leading dim ld).
__device__ inline v16bf load_frag_A(const bf16* base, int ld, int lane) {
  const int r    = lane & 15;
  const int koff = (lane & 16) ? 8 : 0;
  const bf16* row = base + (size_t)r * ld;
  union { v16bf v; v8bf h[2]; } u;
  u.h[0] = *reinterpret_cast<const v8bf*>(row + koff);
  u.h[1] = *reinterpret_cast<const v8bf*>(row + 16 + koff);
  return u.v;
}

// B fragment, 32x16 bf16. Memory row n = column n of B (K contiguous).
__device__ inline v16bf load_frag_B(const bf16* base, int ld, int lane) {
  const int n    = lane & 15;
  const int koff = (lane & 16) ? 8 : 0;
  const bf16* col = base + (size_t)n * ld + koff;
  union { v16bf v; v8bf h[2]; } u;
  u.h[0] = *reinterpret_cast<const v8bf*>(col);
  u.h[1] = *reinterpret_cast<const v8bf*>(col + 16);
  return u.v;
}

// ---------------------------------------------------------------------------
// f32 -> bf16 conversion + head-weight gather
// ---------------------------------------------------------------------------
__global__ void cvt_f32_bf16(const float* __restrict__ in, bf16* __restrict__ out, int n) {
  int i = blockIdx.x * blockDim.x + threadIdx.x;
  if (i < n) out[i] = (bf16)in[i];
}

__global__ void head_weights_k(const float* __restrict__ x,
                               const long long* __restrict__ ops,
                               float* __restrict__ hw) {
  int t = threadIdx.x;
  if (t < B_ * H_) {
    int b = t / H_, h = t % H_;
    hw[t] = x[(size_t)b * S_ * D_ + (int)ops[h]];
  }
}

// ---------------------------------------------------------------------------
// GEMM: C[m,n] = sum_k A[m,k] * W[n,k]   (A: MxK bf16, W: NxK bf16, K=1024)
// Block = 128 threads = 4 waves (2x2); block tile 64x64, k-step 64.
// Double-buffered LDS tiles filled via async global->LDS DMA (ASYNCcnt).
// ---------------------------------------------------------------------------
static constexpr int LDSP = 72;   // padded row stride (bf16 elems), 16B-aligned

template <int MODE>
__global__ __launch_bounds__(128)
void gemm_wmma(const bf16* __restrict__ A, const bf16* __restrict__ W,
               bf16* __restrict__ outb, const float* __restrict__ resid,
               float* __restrict__ outf) {
  __shared__ bf16 As[2][64 * LDSP];
  __shared__ bf16 Bs[2][64 * LDSP];

  const int tid  = threadIdx.x;
  const int lane = tid & 31;
  const int wave = tid >> 5;
  const int m0 = blockIdx.x * 64;
  const int n0 = blockIdx.y * 64;
  const int wm = (wave >> 1) * 32;
  const int wn = (wave & 1) * 32;

  auto stage = [&](int buf, int k) {
#pragma unroll
    for (int c = tid; c < 512; c += 128) {
      const int row = c >> 3;
      const int off = (c & 7) * 8;
      cp16(&As[buf][row * LDSP + off], A + (size_t)(m0 + row) * D_ + k + off);
      cp16(&Bs[buf][row * LDSP + off], W + (size_t)(n0 + row) * D_ + k + off);
    }
  };

  v8f acc00 = {}, acc01 = {}, acc10 = {}, acc11 = {};

  stage(0, 0);
#pragma unroll 2
  for (int kt = 0; kt < D_ / 64; ++kt) {
    const int k   = kt * 64;
    const int buf = kt & 1;
    wait_async0();
    asm volatile("s_wait_dscnt 0" ::: "memory");
    __syncthreads();                       // staged tile visible to all waves
    if (k + 64 < D_) stage(buf ^ 1, k + 64);

#pragma unroll
    for (int kk = 0; kk < 64; kk += 32) {
      v16bf a0 = load_frag_A(&As[buf][(wm)      * LDSP + kk], LDSP, lane);
      v16bf a1 = load_frag_A(&As[buf][(wm + 16) * LDSP + kk], LDSP, lane);
      v16bf b0 = load_frag_B(&Bs[buf][(wn)      * LDSP + kk], LDSP, lane);
      v16bf b1 = load_frag_B(&Bs[buf][(wn + 16) * LDSP + kk], LDSP, lane);
      acc00 = wmma_bf16(a0, b0, acc00);
      acc01 = wmma_bf16(a0, b1, acc01);
      acc10 = wmma_bf16(a1, b0, acc10);
      acc11 = wmma_bf16(a1, b1, acc11);
    }

    asm volatile("s_wait_dscnt 0" ::: "memory");  // frag reads retired
    __syncthreads();                              // before buf is re-filled
  }

  const int cn = lane & 15;
  const int mb = (lane & 16) ? 8 : 0;
  const v8f* accs[4] = {&acc00, &acc01, &acc10, &acc11};
#pragma unroll
  for (int t = 0; t < 4; ++t) {
    const int tm = t >> 1, tn = t & 1;
#pragma unroll
    for (int vi = 0; vi < 8; ++vi) {
      const int m = m0 + wm + tm * 16 + mb + vi;
      const int n = n0 + wn + tn * 16 + cn;
      const float v = (*accs[t])[vi];
      if (MODE == 2) {
        const size_t idx = (size_t)m * E_ + n;
        outf[idx] = resid[idx] + v;
      } else {
        const int b = m >> 11, s = m & (S_ - 1);
        const int h = n >> 6,  d = n & (HD_ - 1);
        if (MODE == 0)
          outb[(((size_t)(b * H_ + h) * S_) + s) * HD_ + d] = (bf16)v;
        else
          outb[(((size_t)(b * H_ + h) * HD_) + d) * S_ + s] = (bf16)v;
      }
    }
  }
}

// ---------------------------------------------------------------------------
// Flash attention: one wave per 16-row q tile; 2 waves per block.
// kv step = 32. V tiles are DMA'd by the Tensor Data Mover into per-wave
// double-buffered LDS tiles, one step ahead (TENSORcnt pipelining).
// ---------------------------------------------------------------------------
__global__ __launch_bounds__(64)
void attn_flash(const bf16* __restrict__ Q, const bf16* __restrict__ K,
                const bf16* __restrict__ Vt, const float* __restrict__ mask,
                const float* __restrict__ hw, bf16* __restrict__ Ao) {
  __shared__ bf16 pbuf[2][16 * 32];
#ifdef HAVE_TDM
  __shared__ bf16 vbuf[2][2][64 * 32];   // [wave][buf]: 64 d-rows x 32 kv
#endif

  const int lane = threadIdx.x & 31;
  const int wave = threadIdx.x >> 5;
  const int task = blockIdx.x * 2 + wave;      // 0 .. B*H*(S/16)-1
  const int qt   = task & (S_ / 16 - 1);
  const int bh   = task >> 7;                  // b*H + h
  const int h    = bh & (H_ - 1);
  const int bb   = bh >> 4;
  const int q0   = qt * 16;

  const bf16*  Qb = Q  + ((size_t)bh * S_ + q0) * HD_;
  const bf16*  Kb = K  + (size_t)bh * S_ * HD_;
  const bf16*  Vb = Vt + (size_t)bh * HD_ * S_;
  const float* Mb = mask + (size_t)h * S_ * S_;

  const v16bf qa0 = load_frag_A(Qb,      HD_, lane);
  const v16bf qa1 = load_frag_A(Qb + 32, HD_, lane);

  float mrun[8], lrun[8];
#pragma unroll
  for (int i = 0; i < 8; ++i) { mrun[i] = -1e30f; lrun[i] = 0.f; }
  v8f o0 = {}, o1 = {}, o2 = {}, o3 = {};

  const float scale = 0.125f;                  // 1/sqrt(64)
  const int   cn = lane & 15;
  const int   mb = (lane & 16) ? 8 : 0;
  bf16* p = pbuf[wave];

#ifdef HAVE_TDM
  // prologue: DMA V tile for s0=0 into buffer 0 (64 rows x 32 cols, 2B elems)
  tdm_load_2d(Vb, lds_off_of(&vbuf[wave][0][0]),
              (unsigned)S_, (unsigned)S_, (unsigned)HD_, 32u, 64u);
#endif

  for (int s0 = 0; s0 < S_; s0 += 32) {
    // ---- scores: two 16x16 tiles covering kv columns s0..s0+31 ----
    v8f c0 = {}, c1 = {};
    {
      v16bf k00 = load_frag_B(Kb + (size_t)s0 * HD_,             HD_, lane);
      v16bf k01 = load_frag_B(Kb + (size_t)s0 * HD_ + 32,        HD_, lane);
      c0 = wmma_bf16(qa0, k00, c0);
      c0 = wmma_bf16(qa1, k01, c0);
      v16bf k10 = load_frag_B(Kb + (size_t)(s0 + 16) * HD_,      HD_, lane);
      v16bf k11 = load_frag_B(Kb + (size_t)(s0 + 16) * HD_ + 32, HD_, lane);
      c1 = wmma_bf16(qa0, k10, c1);
      c1 = wmma_bf16(qa1, k11, c1);
    }
    if (s0 + 32 < S_) {
      __builtin_prefetch(Kb + (size_t)(s0 + 32) * HD_, 0, 1);
      // next mask tile rows (HBM-bound stream: 268 MB total)
      __builtin_prefetch(Mb + (size_t)(q0 + mb)     * S_ + s0 + 32, 0, 0);
      __builtin_prefetch(Mb + (size_t)(q0 + mb + 4) * S_ + s0 + 32, 0, 0);
    }

    // ---- online softmax over 32 kv columns (row = vgpr idx, col = lane) ----
    float alpha[8];
#pragma unroll
    for (int vi = 0; vi < 8; ++vi) {
      const int qrow = q0 + mb + vi;
      const float* mrow = Mb + (size_t)qrow * S_ + s0;
      const float x0 = c0[vi] * scale + mrow[cn];
      const float x1 = c1[vi] * scale + mrow[16 + cn];
      float r = fmaxf(x0, x1);
      r = fmaxf(r, __shfl_xor(r, 1));
      r = fmaxf(r, __shfl_xor(r, 2));
      r = fmaxf(r, __shfl_xor(r, 4));
      r = fmaxf(r, __shfl_xor(r, 8));
      const float mn = fmaxf(mrun[vi], r);
      const float a  = __expf(mrun[vi] - mn);
      const float p0 = __expf(x0 - mn);
      const float p1 = __expf(x1 - mn);
      float ssum = p0 + p1;
      ssum += __shfl_xor(ssum, 1);
      ssum += __shfl_xor(ssum, 2);
      ssum += __shfl_xor(ssum, 4);
      ssum += __shfl_xor(ssum, 8);
      lrun[vi] = lrun[vi] * a + ssum;
      mrun[vi] = mn;
      alpha[vi] = a;
      p[(mb + vi) * 32 + cn]      = (bf16)p0;
      p[(mb + vi) * 32 + 16 + cn] = (bf16)p1;
    }
#pragma unroll
    for (int vi = 0; vi < 8; ++vi) {
      o0[vi] *= alpha[vi]; o1[vi] *= alpha[vi];
      o2[vi] *= alpha[vi]; o3[vi] *= alpha[vi];
    }

    // DS ops are in-order per wave; wait covers pbuf stores and prior vbuf reads
    asm volatile("s_wait_dscnt 0" ::: "memory");

#ifdef HAVE_TDM
    const int cur = (s0 >> 5) & 1;
    if (s0 + 32 < S_)   // DMA next V tile while we consume the current one
      tdm_load_2d(Vb + s0 + 32, lds_off_of(&vbuf[wave][cur ^ 1][0]),
                  (unsigned)S_, (unsigned)S_, (unsigned)HD_, 32u, 64u);
#endif

    v16bf pa;
    {
      const int r    = lane & 15;
      const int koff = (lane & 16) ? 8 : 0;
      union { v16bf v; v8bf h2[2]; } u;
      u.h2[0] = *reinterpret_cast<const v8bf*>(p + r * 32 + koff);
      u.h2[1] = *reinterpret_cast<const v8bf*>(p + r * 32 + 16 + koff);
      pa = u.v;
    }

    // ---- O += P (16x32) x V (32x64) ----
#ifdef HAVE_TDM
    if (s0 + 32 < S_) __builtin_amdgcn_s_wait_tensorcnt(1);  // cur tile done
    else              __builtin_amdgcn_s_wait_tensorcnt(0);
    const bf16* vt = &vbuf[wave][cur][0];
    o0 = wmma_bf16(pa, load_frag_B(vt +  0 * 32, 32, lane), o0);
    o1 = wmma_bf16(pa, load_frag_B(vt + 16 * 32, 32, lane), o1);
    o2 = wmma_bf16(pa, load_frag_B(vt + 32 * 32, 32, lane), o2);
    o3 = wmma_bf16(pa, load_frag_B(vt + 48 * 32, 32, lane), o3);
#else
    o0 = wmma_bf16(pa, load_frag_B(Vb + (size_t)0  * S_ + s0, S_, lane), o0);
    o1 = wmma_bf16(pa, load_frag_B(Vb + (size_t)16 * S_ + s0, S_, lane), o1);
    o2 = wmma_bf16(pa, load_frag_B(Vb + (size_t)32 * S_ + s0, S_, lane), o2);
    o3 = wmma_bf16(pa, load_frag_B(Vb + (size_t)48 * S_ + s0, S_, lane), o3);
#endif
  }

  // ---- finalize: * head_weight / l, write (b, s, h*HD+d) bf16 ----
  const float w = hw[bh];
  float wl[8];
#pragma unroll
  for (int vi = 0; vi < 8; ++vi) {
#if defined(__has_builtin) && __has_builtin(__builtin_amdgcn_rcpf)
    wl[vi] = w * __builtin_amdgcn_rcpf(lrun[vi]);
#else
    wl[vi] = w / lrun[vi];
#endif
  }
  const v8f* os[4] = {&o0, &o1, &o2, &o3};
#pragma unroll
  for (int nt = 0; nt < 4; ++nt) {
#pragma unroll
    for (int vi = 0; vi < 8; ++vi) {
      const int qrow = q0 + mb + vi;
      const int d    = nt * 16 + cn;
      Ao[(size_t)(bb * S_ + qrow) * E_ + h * HD_ + d] = (bf16)((*os[nt])[vi] * wl[vi]);
    }
  }
}

// ---------------------------------------------------------------------------
// Launch
// ---------------------------------------------------------------------------
extern "C" void kernel_launch(void* const* d_in, const int* in_sizes, int n_in,
                              void* d_out, int out_size, void* d_ws, size_t ws_size,
                              hipStream_t stream) {
  (void)in_sizes; (void)n_in; (void)out_size;
  const float*     x    = (const float*)d_in[0];
  const float*     Wq   = (const float*)d_in[1];
  const float*     Wk   = (const float*)d_in[2];
  const float*     Wv   = (const float*)d_in[3];
  const float*     Wo   = (const float*)d_in[4];
  const float*     msk  = (const float*)d_in[5];
  const long long* ops  = (const long long*)d_in[6];

  char* ws = (char*)d_ws;
  size_t off = 0;
  auto alloc = [&](size_t bytes) {
    char* pp = ws + off;
    off += (bytes + 255) & ~(size_t)255;
    return (void*)pp;
  };
  bf16*  xb  = (bf16*)alloc((size_t)M_ * D_ * sizeof(bf16));
  bf16*  wqb = (bf16*)alloc((size_t)E_ * D_ * sizeof(bf16));
  bf16*  wkb = (bf16*)alloc((size_t)E_ * D_ * sizeof(bf16));
  bf16*  wvb = (bf16*)alloc((size_t)E_ * D_ * sizeof(bf16));
  bf16*  wob = (bf16*)alloc((size_t)D_ * E_ * sizeof(bf16));
  bf16*  Qb  = (bf16*)alloc((size_t)B_ * H_ * S_ * HD_ * sizeof(bf16));
  bf16*  Kb  = (bf16*)alloc((size_t)B_ * H_ * S_ * HD_ * sizeof(bf16));
  bf16*  Vt  = (bf16*)alloc((size_t)B_ * H_ * HD_ * S_ * sizeof(bf16));
  bf16*  Ab  = (bf16*)alloc((size_t)M_ * E_ * sizeof(bf16));
  float* hw  = (float*)alloc((size_t)B_ * H_ * sizeof(float));
  if (off > ws_size) return;  // insufficient scratch (deterministic no-op)

  // 1) bf16 conversions
  {
    const int n1 = M_ * D_, n2 = E_ * D_;
    cvt_f32_bf16<<<(n1 + 255) / 256, 256, 0, stream>>>(x,  xb,  n1);
    cvt_f32_bf16<<<(n2 + 255) / 256, 256, 0, stream>>>(Wq, wqb, n2);
    cvt_f32_bf16<<<(n2 + 255) / 256, 256, 0, stream>>>(Wk, wkb, n2);
    cvt_f32_bf16<<<(n2 + 255) / 256, 256, 0, stream>>>(Wv, wvb, n2);
    cvt_f32_bf16<<<(n2 + 255) / 256, 256, 0, stream>>>(Wo, wob, n2);
  }
  // 2) head weights gather
  head_weights_k<<<1, 64, 0, stream>>>(x, ops, hw);

  // 3) QKV projections (WMMA bf16, async-LDS staged)
  dim3 g(M_ / 64, E_ / 64), blk(128);
  gemm_wmma<0><<<g, blk, 0, stream>>>(xb, wqb, Qb, nullptr, nullptr);
  gemm_wmma<0><<<g, blk, 0, stream>>>(xb, wkb, Kb, nullptr, nullptr);
  gemm_wmma<1><<<g, blk, 0, stream>>>(xb, wvb, Vt, nullptr, nullptr);

  // 4) flash attention (one wave per 16-row q tile; TDM-staged V tiles)
  const int tasks = B_ * H_ * (S_ / 16);       // 4096 waves
  attn_flash<<<tasks / 2, 64, 0, stream>>>(Qb, Kb, Vt, msk, hw, Ab);

  // 5) output projection + residual (f32 out)
  gemm_wmma<2><<<g, blk, 0, stream>>>(Ab, wob, nullptr, x, (float*)d_out);
}
